// UnICORNN_22686017257635
// MI455X (gfx1250) — matrix operand order repaired and checked by reading, not compiled, not run on hardware
//
#include <hip/hip_runtime.h>

typedef __attribute__((ext_vector_type(16))) _Float16 v16h;
typedef __attribute__((ext_vector_type(8)))  _Float16 v8h;
typedef __attribute__((ext_vector_type(8)))  float    v8f;

#define HID     256
#define IN_DIM  128
#define OUT_DIM 128
#define N_LAYERS 2
#define DT_C    0.03f
#define ALPHA_C 0.9f

// Fragment-linear state layout: [tile (bi*16+bj)][lane 0..31][v 0..7] floats.
// One lane's 8 state values = 32 contiguous bytes -> 2x global b128.
__device__ __forceinline__ int ws_off(int tile, int lane) { return (tile * 32 + lane) * 8; }

// ---------------------------------------------------------------------------
// Phase 0a: h = x @ w_in.T + b_in, written DIRECTLY in fragment-linear layout.
// C-fragment mapping: tile(bi=i>>4, bj=j>>4), lane=((i>>3)&1)*16 + (j&15), v=i&7
// ---------------------------------------------------------------------------
__global__ void input_linear_kernel(const float* __restrict__ x,
                                    const float* __restrict__ w_in,
                                    const float* __restrict__ b_in,
                                    float* __restrict__ x_ws) {
    const int i = blockIdx.x;     // batch row 0..255
    const int j = threadIdx.x;    // hidden col 0..255
    const float* xr = x    + i * IN_DIM;
    const float* wr = w_in + j * IN_DIM;
    float acc = b_in[j];
    #pragma unroll 8
    for (int d = 0; d < IN_DIM; ++d) acc = fmaf(xr[d], wr[d], acc);
    const int tile = (i >> 4) * 16 + (j >> 4);
    const int lane = ((i >> 3) & 1) * 16 + (j & 15);
    x_ws[ws_off(tile, lane) + (i & 7)] = acc;
}

// ---------------------------------------------------------------------------
// Phase 0b: v_rec fp32 -> f16 (row-major; A-fragments load straight from it)
// ---------------------------------------------------------------------------
__global__ void cvt_v_f16_kernel(const float* __restrict__ v,
                                 _Float16* __restrict__ vh) {
    const int i = blockIdx.x * 256 + threadIdx.x;
    vh[i] = (_Float16)v[i];
}

__device__ __forceinline__ float fast_tanh(float p) {
    float e = __expf(2.0f * p);
    return 1.0f - 2.0f / (e + 1.0f);
}

// ---------------------------------------------------------------------------
// Persistent UnICORNN recurrence. One 512-thread workgroup (16 waves).
//
// Wave w owns row blocks {biL=w&7, biH=biL+8} x 8 column tiles bj0=(w>>3)*8.
// x,z fp32 master state streams through L2 (fragment-linear ws).
// f16 x image lives in LDS in B-fragment-native layout, DOUBLE BUFFERED:
//   xB[buf][kb][bjB][lane][16 halves]; matmuls read buf=cur, the elementwise
//   phase scatters updated x into cur^1 -> exactly one s_barrier per step.
// A operands: two global_load_b128 per lane from row-major f16 V.
// ---------------------------------------------------------------------------
__global__ void __launch_bounds__(512, 1)
unicornn_persist_kernel(const _Float16* __restrict__ vf16,   // [2,256,256] f16
                        const float* __restrict__ w_rec,     // [2,256]
                        const float* __restrict__ b_rec,     // [2,256]
                        float* __restrict__ x_ws,            // [256][32][8] f32
                        float* __restrict__ z_ws,            // [256][32][8] f32
                        float* __restrict__ h_out) {         // [256,256] f32 row-major
    extern __shared__ __align__(32) _Float16 xB[];  // 2 * 128 frags * 32 lanes * 16 halves

    const int tid  = threadIdx.x;
    const int lane = tid & 31;
    const int wave = tid >> 5;          // 0..15
    const int biL  = wave & 7;
    const int biH  = biL + 8;
    const int bj0  = (wave >> 3) * 8;
    const int g    = lane >> 4;
    const int m    = lane & 15;

    const v8f vzero = {0.f, 0.f, 0.f, 0.f, 0.f, 0.f, 0.f, 0.f};

    // ---- prologue: publish x (f16) into LDS buffer 0 ----
    #pragma unroll
    for (int s = 0; s < 2; ++s) {
        const int bi2 = s ? biH : biL;
        #pragma unroll
        for (int t = 0; t < 8; ++t) {
            const int bj = bj0 + t;
            const v8f xv = *(const v8f*)(x_ws + ws_off(bi2 * 16 + bj, lane));
            const int fb = ((bj >> 1) * 16 + bi2) * 32;   // buf 0
            #pragma unroll
            for (int v = 0; v < 8; ++v)
                xB[(fb + ((bj & 1) * 16 + v + 8 * g)) * 16 + m] = (_Float16)xv[v];
        }
    }
    __syncthreads();

    int cur = 0;
    for (int layer = 0; layer < N_LAYERS; ++layer) {
        // per-lane elementwise params for this wave's 8 columns
        float Wv[8], Bv[8];
        #pragma unroll
        for (int t = 0; t < 8; ++t) {
            const int j = (bj0 + t) * 16 + m;
            Wv[t] = w_rec[layer * HID + j];
            Bv[t] = b_rec[layer * HID + j];
        }
        // z resets to zero at layer start (owned tiles only; no cross-wave dep)
        #pragma unroll
        for (int s = 0; s < 2; ++s)
            #pragma unroll
            for (int t = 0; t < 8; ++t)
                *(v8f*)(z_ws + ws_off((s ? biH : biL) * 16 + bj0 + t, lane)) = vzero;

        const _Float16* Vl = vf16 + layer * HID * HID;

        for (int step = 0; step < HID; ++step) {
            const int nxt = cur ^ 1;
            #pragma unroll
            for (int g2 = 0; g2 < 2; ++g2) {
                // ---- matmul: 8 tiles (2 bi x 4 bj), K = 256 in 8 chunks ----
                v8f acc[2][4];
                #pragma unroll
                for (int s = 0; s < 2; ++s)
                    #pragma unroll
                    for (int t = 0; t < 4; ++t) acc[s][t] = vzero;

                #pragma unroll
                for (int kb = 0; kb < 8; ++kb) {
                    v16h Alo, Ahi;
                    {
                        const _Float16* ar = Vl + (biL * 16 + m) * HID + kb * 32;
                        ((v8h*)&Alo)[0] = *(const v8h*)(ar + 16 * g);
                        ((v8h*)&Alo)[1] = *(const v8h*)(ar + 16 * g + 8);
                    }
                    {
                        const _Float16* ar = Vl + (biH * 16 + m) * HID + kb * 32;
                        ((v8h*)&Ahi)[0] = *(const v8h*)(ar + 16 * g);
                        ((v8h*)&Ahi)[1] = *(const v8h*)(ar + 16 * g + 8);
                    }
                    #pragma unroll
                    for (int t = 0; t < 4; ++t) {
                        const int bj = bj0 + g2 * 4 + t;
                        const v16h Bf =
                            *(const v16h*)(xB + ((cur * 128 + kb * 16 + bj) * 32 + lane) * 16);
                        acc[0][t] = __builtin_amdgcn_wmma_f32_16x16x32_f16(
                            false, Alo, false, Bf, (short)0, acc[0][t], false, false);
                        acc[1][t] = __builtin_amdgcn_wmma_f32_16x16x32_f16(
                            false, Ahi, false, Bf, (short)0, acc[1][t], false, false);
                    }
                }

                // ---- elementwise oscillator update; scatter new x into nxt ----
                #pragma unroll
                for (int s = 0; s < 2; ++s) {
                    const int bi2 = s ? biH : biL;
                    #pragma unroll
                    for (int t = 0; t < 4; ++t) {
                        const int bj = bj0 + g2 * 4 + t;
                        const int tn = bi2 * 16 + bj;
                        v8f xv = *(const v8f*)(x_ws + ws_off(tn, lane));
                        v8f zv = *(const v8f*)(z_ws + ws_off(tn, lane));
                        const float Wj = Wv[g2 * 4 + t];
                        const float Bj = Bv[g2 * 4 + t];
                        #pragma unroll
                        for (int v = 0; v < 8; ++v) {
                            const float pre = fmaf(Wj, xv[v], acc[s][t][v] + Bj);
                            const float th  = fast_tanh(pre);
                            zv[v] -= DT_C * (th + ALPHA_C * xv[v]);
                            xv[v] += DT_C * zv[v];
                        }
                        *(v8f*)(x_ws + ws_off(tn, lane)) = xv;
                        *(v8f*)(z_ws + ws_off(tn, lane)) = zv;
                        const int fb = (nxt * 128 + (bj >> 1) * 16 + bi2) * 32;
                        #pragma unroll
                        for (int v = 0; v < 8; ++v)
                            xB[(fb + ((bj & 1) * 16 + v + 8 * g)) * 16 + m] = (_Float16)xv[v];
                    }
                }
            }
            __syncthreads();   // everyone done reading cur / writing nxt
            cur = nxt;
        }
    }

    // ---- epilogue: final x -> h_out, row-major ----
    #pragma unroll
    for (int s = 0; s < 2; ++s) {
        const int bi2 = s ? biH : biL;
        #pragma unroll
        for (int t = 0; t < 8; ++t) {
            const int bj = bj0 + t;
            const v8f xv = *(const v8f*)(x_ws + ws_off(bi2 * 16 + bj, lane));
            #pragma unroll
            for (int v = 0; v < 8; ++v) {
                const int i = bi2 * 16 + v + 8 * g;
                h_out[i * HID + bj * 16 + m] = xv[v];
            }
        }
    }
}

// ---------------------------------------------------------------------------
// Phase 2: out = h @ w_out.T + b_out
// ---------------------------------------------------------------------------
__global__ void output_linear_kernel(const float* __restrict__ hfin,
                                     const float* __restrict__ w_out,
                                     const float* __restrict__ b_out,
                                     float* __restrict__ out) {
    const int row = blockIdx.x;    // 0..255
    const int col = threadIdx.x;   // 0..127
    const float* hr = hfin  + row * HID;
    const float* wr = w_out + col * HID;
    float acc = b_out[col];
    #pragma unroll 8
    for (int d = 0; d < HID; ++d) acc = fmaf(hr[d], wr[d], acc);
    out[row * OUT_DIM + col] = acc;
}

// ---------------------------------------------------------------------------
extern "C" void kernel_launch(void* const* d_in, const int* in_sizes, int n_in,
                              void* d_out, int out_size, void* d_ws, size_t ws_size,
                              hipStream_t stream) {
    (void)in_sizes; (void)n_in; (void)out_size; (void)ws_size;
    const float* x     = (const float*)d_in[0];
    const float* w_in  = (const float*)d_in[1];
    const float* b_in  = (const float*)d_in[2];
    const float* w_rec = (const float*)d_in[3];
    const float* v_rec = (const float*)d_in[4];
    const float* b_rec = (const float*)d_in[5];
    const float* w_out = (const float*)d_in[6];
    const float* b_out = (const float*)d_in[7];
    float* out = (float*)d_out;

    char* ws = (char*)d_ws;
    float*    x_ws   = (float*)   (ws);                 // 256 KB, fragment-linear x (fp32)
    float*    z_ws   = (float*)   (ws + 256 * 1024);    // 256 KB, fragment-linear z (fp32)
    _Float16* vf16   = (_Float16*)(ws + 512 * 1024);    // 256 KB, V (2 layers) f16
    float*    hfinal = (float*)   (ws + 768 * 1024);    // 256 KB, final h row-major

    input_linear_kernel<<<HID, HID, 0, stream>>>(x, w_in, b_in, x_ws);
    cvt_v_f16_kernel<<<(N_LAYERS * HID * HID) / 256, 256, 0, stream>>>(v_rec, vf16);
    unicornn_persist_kernel<<<1, 512, 256 * 1024, stream>>>(vf16, w_rec, b_rec,
                                                            x_ws, z_ws, hfinal);
    output_linear_kernel<<<HID, OUT_DIM, 0, stream>>>(hfinal, w_out, b_out, out);
}
